// FCOS_16801912062473
// MI455X (gfx1250) — matrix operand order repaired and checked by reading, not compiled, not run on hardware
//
#include <hip/hip_runtime.h>

typedef __attribute__((ext_vector_type(16))) __bf16 v16bf;
typedef __attribute__((ext_vector_type(8)))  __bf16 v8bf;
typedef __attribute__((ext_vector_type(8)))  float  v8f;

#define TPB 256

// ---------------- zero page init (for branch-free zero padding) ----------------
__global__ __launch_bounds__(TPB) void zero_page_kernel(uint4* __restrict__ z) {
  if (threadIdx.x < 64) z[threadIdx.x] = make_uint4(0u, 0u, 0u, 0u);  // 1 KB
}

// ---------------- fp32 NCHW -> bf16 NHWC ----------------
__global__ __launch_bounds__(TPB) void f2bf_nhwc_kernel(const float* __restrict__ in,
                                                        __bf16* __restrict__ out,
                                                        int HW, int total) {
  int i = blockIdx.x * blockDim.x + threadIdx.x;  // NHWC flat index
  if (i >= total) return;
  int c = i & 255;
  int q = i >> 8;          // n*HW + p
  int n = q / HW;
  int p = q - n * HW;
  out[i] = (__bf16)in[((size_t)n * 256 + c) * HW + p];
}

// ---------------- fp32 [O,256,3,3] -> bf16 tap-major [9][O][256] ----------------
__global__ __launch_bounds__(TPB) void pack_w_kernel(const float* __restrict__ in,
                                                     __bf16* __restrict__ out, int O,
                                                     int total) {
  int i = blockIdx.x * blockDim.x + threadIdx.x;  // (tap*O + o)*256 + ci
  if (i >= total) return;
  int ci = i & 255;
  int r  = i >> 8;
  int o  = r % O;
  int tap = r / O;
  out[i] = (__bf16)in[((size_t)o * 256 + ci) * 9 + tap];
}

// ---------------- implicit-GEMM 3x3 conv via WMMA bf16 ----------------
// X:   NHWC bf16 [2*H*W][256];  Wpk: tap-major bf16 [9][Oreal][256]
// Block tile 64 out-ch x 64 px, 8 waves (4 co x 2 px blocks), 2 wmma/wave/K-step.
// LDS double-buffered; zero-page select instead of predicated loads.
template <int NHWC_OUT>
__global__ __launch_bounds__(TPB) void conv3x3_wmma_kernel(
    const __bf16* __restrict__ X, const __bf16* __restrict__ Wpk,
    const float* __restrict__ bias, float* __restrict__ Y,
    const __bf16* __restrict__ zpage,
    int H, int W, int Oreal, int act) {
  const int HWp    = H * W;
  const int P      = 2 * HWp;
  const int pBase  = blockIdx.x * 64;
  const int coBase = blockIdx.y * 64;

  __shared__ __align__(16) __bf16 sA[2][64][40];   // [buf][out-ch][cin]
  __shared__ __align__(16) __bf16 sBt[2][64][40];  // [buf][pixel][cin]

  const int tid  = threadIdx.x;
  const int lane = tid & 31;
  const int wave = tid >> 5;
  const int wi = wave & 3;   // 16-row (out-channel) block within 64
  const int wj = wave >> 2;  // 32-col (pixel) block within 64

  // Staging decode: one b128 (8 cin) per thread per matrix per K-step.
  const int sp = tid >> 2;       // 0..63 : A row / B pixel
  const int sc = (tid & 3) * 8;  // 0,8,16,24 : cin chunk
  const int pgS = pBase + sp;
  const bool pv = pgS < P;
  int pn = 0, pyv = 0, pxv = 0;
  if (pv) {
    pn = pgS / HWp;
    int rem = pgS - pn * HWp;
    pyv = rem / W;
    pxv = rem - pyv * W;
  }
  const int oS = coBase + sp;
  // Weight rows beyond Oreal read the zero page (branch-free).
  const bool aOK = oS < Oreal;

  v8f acc0 = {}, acc1 = {};

  const int m    = lane & 15;
  const int hi8  = (lane >> 4) ? 16 : 0;   // A K-base
  const int hi16 = (lane >> 4) ? 16 : 0;   // B K-base
  const int rowA = wi * 16 + m;
  const int col0 = wj * 32 + m;
  const int col1 = col0 + 16;

  int buf = 0;
#pragma unroll
  for (int tap = 0; tap < 9; ++tap) {
    const int dy = tap / 3 - 1;      // compile-time after unroll
    const int dx = tap % 3 - 1;
    const int sy = pyv + dy;
    const int sx = pxv + dx;
    const bool bvld = pv && (unsigned)sy < (unsigned)H && (unsigned)sx < (unsigned)W;
    // Branch-free zero padding: OOB taps stream from the zeroed page.
    const __bf16* bBase = bvld ? &X[((size_t)(pn * HWp + sy * W + sx)) << 8] : zpage;
    const __bf16* aBase = aOK ? &Wpk[((size_t)tap * Oreal + oS) << 8] : zpage;

    // Prologue: stage ck=0 of this tap.
    {
      uint4 av = *(const uint4*)&aBase[sc];
      uint4 bv = *(const uint4*)&bBase[sc];
      *(uint4*)&sA[buf][sp][sc]  = av;
      *(uint4*)&sBt[buf][sp][sc] = bv;
    }
    __syncthreads();

#pragma unroll
    for (int ck = 0; ck < 256; ck += 32) {
      const bool hasNext = (ck + 32) < 256;
      // Issue next chunk's global loads first so they overlap the WMMAs.
      uint4 avN, bvN;
      if (hasNext) {
        avN = *(const uint4*)&aBase[ck + 32 + sc];
        bvN = *(const uint4*)&bBase[ck + 32 + sc];
      }

      // Fragments in contiguous k order (k = 16*half + s), same for A and B -- all ds_load_b128.
      v8bf a0 = *(const v8bf*)&sA[buf][rowA][hi8];
      v8bf a1 = *(const v8bf*)&sA[buf][rowA][hi8 + 8];
      v16bf a = __builtin_shufflevector(a0, a1, 0, 1, 2, 3, 4, 5, 6, 7,
                                        8, 9, 10, 11, 12, 13, 14, 15);
      v8bf b0a = *(const v8bf*)&sBt[buf][col0][hi16];
      v8bf b0b = *(const v8bf*)&sBt[buf][col0][hi16 + 8];
      v16bf b0 = __builtin_shufflevector(b0a, b0b, 0, 1, 2, 3, 4, 5, 6, 7,
                                         8, 9, 10, 11, 12, 13, 14, 15);
      v8bf b1a = *(const v8bf*)&sBt[buf][col1][hi16];
      v8bf b1b = *(const v8bf*)&sBt[buf][col1][hi16 + 8];
      v16bf b1 = __builtin_shufflevector(b1a, b1b, 0, 1, 2, 3, 4, 5, 6, 7,
                                         8, 9, 10, 11, 12, 13, 14, 15);
      acc0 = __builtin_amdgcn_wmma_f32_16x16x32_bf16(false, a, false, b0,
                                                     (short)0, acc0, false, false);
      acc1 = __builtin_amdgcn_wmma_f32_16x16x32_bf16(false, a, false, b1,
                                                     (short)0, acc1, false, false);

      // Store next chunk into the alternate buffer; single barrier per step.
      if (hasNext) {
        *(uint4*)&sA[buf ^ 1][sp][sc]  = avN;
        *(uint4*)&sBt[buf ^ 1][sp][sc] = bvN;
      }
      __syncthreads();
      buf ^= 1;
    }
  }

  // Epilogue. C/D layout: lane holds N=lane&15, VGPR v -> M = v + (lane>=16 ? 8 : 0)
  const int nCol = lane & 15;
  const int mHi  = (lane >> 4) ? 8 : 0;
  const int co0  = coBase + wi * 16 + mHi;  // 8 consecutive out-channels
#pragma unroll
  for (int pj = 0; pj < 2; ++pj) {
    v8f acc = pj ? acc1 : acc0;
    int pg = pBase + wj * 32 + pj * 16 + nCol;
    if (pg >= P) continue;
    if (NHWC_OUT) {
      float4 blo = *(const float4*)&bias[co0];
      float4 bhi = *(const float4*)&bias[co0 + 4];
      float4 v0 = make_float4(acc[0] + blo.x, acc[1] + blo.y,
                              acc[2] + blo.z, acc[3] + blo.w);
      float4 v1 = make_float4(acc[4] + bhi.x, acc[5] + bhi.y,
                              acc[6] + bhi.z, acc[7] + bhi.w);
      if (act) {
        v0.x = fmaxf(v0.x, 0.f); v0.y = fmaxf(v0.y, 0.f);
        v0.z = fmaxf(v0.z, 0.f); v0.w = fmaxf(v0.w, 0.f);
        v1.x = fmaxf(v1.x, 0.f); v1.y = fmaxf(v1.y, 0.f);
        v1.z = fmaxf(v1.z, 0.f); v1.w = fmaxf(v1.w, 0.f);
      }
      size_t baseo = ((size_t)pg << 8) + co0;
      *(float4*)&Y[baseo]     = v0;
      *(float4*)&Y[baseo + 4] = v1;
    } else {
      int n   = pg / HWp;
      int rem = pg - n * HWp;
#pragma unroll
      for (int vv = 0; vv < 8; ++vv) {
        int co = co0 + vv;
        if (co < Oreal) {
          float val = acc[vv] + bias[co];
          if (act) val = fmaxf(val, 0.0f);
          Y[((size_t)n * Oreal + co) * HWp + rem] = val;
        }
      }
    }
  }
}

// ---------------- GroupNorm stats (NHWC): one block per (n, group) ----------------
__global__ __launch_bounds__(TPB) void gn_stats_kernel(const float* __restrict__ X,
                                                       float* __restrict__ stats,
                                                       int HW) {
  __shared__ float ssum[TPB], ssq[TPB];
  int b = blockIdx.x;   // n*16 + g
  int n = b >> 4;
  int g = b & 15;
  int cnt = 16 * HW;
  float s = 0.f, q = 0.f;
  for (int j = threadIdx.x; j < cnt; j += TPB) {
    int p = j >> 4;
    int c = j & 15;
    float v = X[(((size_t)(n * HW + p)) << 8) + g * 16 + c];
    s += v;
    q += v * v;
  }
  ssum[threadIdx.x] = s;
  ssq[threadIdx.x]  = q;
  __syncthreads();
  for (int off = TPB / 2; off > 0; off >>= 1) {
    if (threadIdx.x < off) {
      ssum[threadIdx.x] += ssum[threadIdx.x + off];
      ssq[threadIdx.x]  += ssq[threadIdx.x + off];
    }
    __syncthreads();
  }
  if (threadIdx.x == 0) {
    float mean = ssum[0] / cnt;
    float var  = ssq[0] / cnt - mean * mean;
    stats[b * 2]     = mean;
    stats[b * 2 + 1] = rsqrtf(fmaxf(var, 0.f) + 1e-5f);
  }
}

// ---------------- GN apply + affine + ReLU (NHWC fp32 -> NHWC bf16) ----------------
__global__ __launch_bounds__(TPB) void gn_apply_kernel(
    const float* __restrict__ X, const float* __restrict__ stats,
    const float* __restrict__ gamma, const float* __restrict__ beta,
    __bf16* __restrict__ out, int HW, int total) {
  int i = blockIdx.x * blockDim.x + threadIdx.x;
  if (i >= total) return;
  int c = i & 255;
  int q = i >> 8;
  int n = q / HW;
  int g = c >> 4;
  float mean = stats[(n * 16 + g) * 2];
  float rstd = stats[(n * 16 + g) * 2 + 1];
  float v = (X[i] - mean) * rstd * gamma[c] + beta[c];
  out[i] = (__bf16)fmaxf(v, 0.f);
}

// ---------------- host orchestration ----------------
extern "C" void kernel_launch(void* const* d_in, const int* in_sizes, int n_in,
                              void* d_out, int out_size, void* d_ws, size_t ws_size,
                              hipStream_t stream) {
  (void)in_sizes; (void)n_in; (void)out_size;
  static const int HS[5] = {128, 64, 32, 16, 8};

  // Inputs in setup_inputs() insertion order.
  const float* feat[5];
  for (int i = 0; i < 5; ++i) feat[i] = (const float*)d_in[i];
  const float *cw[2][3], *cb[2][3], *cg[2][3], *cbt[2][3], *fw[2], *fb[2];
  int base = 5;
  for (int t = 0; t < 2; ++t) {
    for (int i = 0; i < 3; ++i) {
      cw[t][i]  = (const float*)d_in[base + 0];
      cb[t][i]  = (const float*)d_in[base + 1];
      cg[t][i]  = (const float*)d_in[base + 2];
      cbt[t][i] = (const float*)d_in[base + 3];
      base += 4;
    }
    fw[t] = (const float*)d_in[base++];
    fb[t] = (const float*)d_in[base++];
  }
  const float* ctrw = (const float*)d_in[base++];
  const float* ctrb = (const float*)d_in[base++];

  // Workspace layout
  char* ws = (char*)d_ws;
  const size_t MB = (size_t)1 << 20;
  if (ws_size < 77 * MB) return;
  __bf16* actA    = (__bf16*)(ws);                    // 16 MB (NHWC bf16)
  __bf16* actB    = (__bf16*)(ws + 17 * MB);          // 16 MB (NHWC bf16)
  float*  convOut = (float*)(ws + 34 * MB);           // 32 MB (NHWC fp32)
  __bf16* wbf     = (__bf16*)(ws + 68 * MB);          // ~7.2 MB (tap-major bf16)
  float*  stats   = (float*)(ws + 76 * MB);           // 256 B
  __bf16* zpage   = (__bf16*)(ws + 76 * MB + 4096);   // 1 KB zeros

  zero_page_kernel<<<1, TPB, 0, stream>>>((uint4*)zpage);

  const int WCONV = 256 * 256 * 9;
  __bf16* wbfConv[2][3];
  __bf16* wbfFin[2];
  __bf16* wbfCtr;
  size_t off = 0;
  for (int t = 0; t < 2; ++t)
    for (int i = 0; i < 3; ++i) { wbfConv[t][i] = wbf + off; off += WCONV; }
  wbfFin[0] = wbf + off; off += 80 * 256 * 9;
  wbfFin[1] = wbf + off; off += 4 * 256 * 9;
  wbfCtr    = wbf + off; off += 256 * 9;

  auto packw = [&](const float* src, __bf16* dst, int O) {
    int n = O * 256 * 9;
    pack_w_kernel<<<(n + TPB - 1) / TPB, TPB, 0, stream>>>(src, dst, O, n);
  };
  for (int t = 0; t < 2; ++t)
    for (int i = 0; i < 3; ++i) packw(cw[t][i], wbfConv[t][i], 256);
  packw(fw[0], wbfFin[0], 80);
  packw(fw[1], wbfFin[1], 4);
  packw(ctrw, wbfCtr, 1);

  // Output offsets: 5x cls, 5x reg, 5x ctr (flattened tuple, NCHW each).
  float* outF = (float*)d_out;
  size_t clsOff[5], regOff[5], ctrOff[5], acc0 = 0;
  for (int L = 0; L < 5; ++L) { clsOff[L] = acc0; acc0 += (size_t)2 * 80 * HS[L] * HS[L]; }
  for (int L = 0; L < 5; ++L) { regOff[L] = acc0; acc0 += (size_t)2 * 4  * HS[L] * HS[L]; }
  for (int L = 0; L < 5; ++L) { ctrOff[L] = acc0; acc0 += (size_t)2 * 1  * HS[L] * HS[L]; }

  for (int L = 0; L < 5; ++L) {
    const int H = HS[L], W = HS[L];
    const int HW = H * W;
    const int P = 2 * HW;
    const int total = 2 * 256 * HW;
    dim3 gTower((P + 63) / 64, 4);       // 256 out channels / 64
    for (int t = 0; t < 2; ++t) {        // t=0 cls tower, t=1 reg tower
      f2bf_nhwc_kernel<<<(total + TPB - 1) / TPB, TPB, 0, stream>>>(feat[L], actA,
                                                                    HW, total);
      __bf16* cur = actA;
      __bf16* nxt = actB;
      for (int i = 0; i < 3; ++i) {
        conv3x3_wmma_kernel<1><<<gTower, TPB, 0, stream>>>(
            cur, wbfConv[t][i], cb[t][i], convOut, zpage, H, W, 256, 0);
        gn_stats_kernel<<<32, TPB, 0, stream>>>(convOut, stats, HW);
        gn_apply_kernel<<<(total + TPB - 1) / TPB, TPB, 0, stream>>>(
            convOut, stats, cg[t][i], cbt[t][i], nxt, HW, total);
        __bf16* tmp = cur; cur = nxt; nxt = tmp;
      }
      if (t == 0) {
        dim3 g((P + 63) / 64, 2);        // 80 out channels -> 2 tiles of 64
        conv3x3_wmma_kernel<0><<<g, TPB, 0, stream>>>(
            cur, wbfFin[0], fb[0], outF + clsOff[L], zpage, H, W, 80, 0);
      } else {
        dim3 g((P + 63) / 64, 1);
        conv3x3_wmma_kernel<0><<<g, TPB, 0, stream>>>(
            cur, wbfFin[1], fb[1], outF + regOff[L], zpage, H, W, 4, 1);
        conv3x3_wmma_kernel<0><<<g, TPB, 0, stream>>>(
            cur, wbfCtr, ctrb, outF + ctrOff[L], zpage, H, W, 1, 0);
      }
    }
  }
}
